// LSTM_11587821764665
// MI455X (gfx1250) — matrix operand order repaired and checked by reading, not compiled, not run on hardware
//
#include <hip/hip_runtime.h>
#include <stdint.h>

// ---------------------------------------------------------------------------
// 2-layer LSTM (B=64, S=512, H=512, D=300) as one persistent wave32 kernel.
// bf16 WMMA 16x16x32 (f32 accum) for the recurrent GEMMs; gates fused with
// sigmoid/tanh state update per wave; 2 device-wide barriers per timestep.
// Nonlinearities use v_exp_f32 + v_rcp_f32 (and v_tanh_f32 when available)
// instead of IEEE divides to shorten the per-step VALU critical path.
// ---------------------------------------------------------------------------

typedef __attribute__((ext_vector_type(16))) __bf16 v16bf;
typedef __attribute__((ext_vector_type(8)))  float  v8f;

#define BATCH 64
#define HSZ   512
#define SEQ   512
#define EMBD  300
#define XPAD  320           // 300 padded to multiple of 32
#define K0LEN 832           // XPAD + HSZ  (layer-0 fused K)
#define K1LEN 1024          // HSZ + HSZ   (layer-1 fused K)
#define G4H   2048          // 4*HSZ gate rows
#define NLBL  10

#define NBLK  16
#define NTHR  256
#define NALL  (NBLK * NTHR) // 4096 threads, 128 waves

union FragU { v16bf bf; uint4 q[2]; };

__device__ __forceinline__ unsigned short f2bf_bits(float f) {
  unsigned u = __float_as_uint(f);
  u += 0x7FFFu + ((u >> 16) & 1u);     // round-to-nearest-even
  return (unsigned short)(u >> 16);
}

__device__ __forceinline__ float fast_rcp(float x) {
  return __builtin_amdgcn_rcpf(x);     // v_rcp_f32 (fast, ~1 ulp)
}
__device__ __forceinline__ float sigm_(float x) {
  return fast_rcp(1.f + __expf(-x));
}
__device__ __forceinline__ float tanh_(float x) {
#if __has_builtin(__builtin_amdgcn_tanhf)
  return __builtin_amdgcn_tanhf(x);    // v_tanh_f32
#else
  return 1.f - 2.f * fast_rcp(__expf(2.f * x) + 1.f);
#endif
}

// A fragment (16-bit A 16x32): lane l holds row M = mBase+(l&15);
// K chunks [kb,kb+8) in v0..3 and [kb+16,kb+24) in v4..7, kb = (l>>4)*8.
__device__ __forceinline__ v16bf load_A(const unsigned short* __restrict__ mat,
                                        int rowlen, int mBase, int k0) {
  const int l = threadIdx.x & 31;
  const unsigned short* p = mat + (size_t)(mBase + (l & 15)) * rowlen + k0 + ((l >> 4) << 3);
  FragU f;
  f.q[0] = *(const uint4*)p;
  f.q[1] = *(const uint4*)(p + 16);
  return f.bf;
}

// B fragment (16-bit B 32x16): lane l holds column N = nBase+(l&15);
// K chunks [kb,kb+8) in v0..3 and [kb+16,kb+24) in v4..7, kb = (l>>4)*8. Source is W[N][K] row-major.
__device__ __forceinline__ v16bf load_B(const unsigned short* __restrict__ mat,
                                        int rowlen, int nBase, int k0) {
  const int l = threadIdx.x & 31;
  const unsigned short* p = mat + (size_t)(nBase + (l & 15)) * rowlen + k0 + ((l >> 4) << 3);
  FragU f;
  f.q[0] = *(const uint4*)p;
  f.q[1] = *(const uint4*)(p + 16);
  return f.bf;
}

// Device-wide sense-reversing barrier (bar[0]=count, bar[1]=generation).
__device__ __forceinline__ void gbar(unsigned* bar) {
  __threadfence();
  __syncthreads();
  if (threadIdx.x == 0) {
    unsigned* cnt = bar;
    unsigned* gen = bar + 1;
    unsigned g = __hip_atomic_load(gen, __ATOMIC_RELAXED, __HIP_MEMORY_SCOPE_AGENT);
    unsigned a = __hip_atomic_fetch_add(cnt, 1u, __ATOMIC_ACQ_REL, __HIP_MEMORY_SCOPE_AGENT);
    if (a == (unsigned)(NBLK - 1)) {
      __hip_atomic_store(cnt, 0u, __ATOMIC_RELAXED, __HIP_MEMORY_SCOPE_AGENT);
      __hip_atomic_fetch_add(gen, 1u, __ATOMIC_RELEASE, __HIP_MEMORY_SCOPE_AGENT);
    } else {
      while (__hip_atomic_load(gen, __ATOMIC_ACQUIRE, __HIP_MEMORY_SCOPE_AGENT) == g)
        __builtin_amdgcn_s_sleep(2);
    }
  }
  __syncthreads();
  __threadfence();   // acquire for all threads of the block
}

// Gather x_t embedding rows (bf16, zero-padded 300->320) into hx0[:,0:320].
__device__ __forceinline__ void gather_x(int tid, int t,
    const int* __restrict__ x, const float* __restrict__ emb,
    unsigned short* __restrict__ hx0) {
  for (int e = tid; e < BATCH * XPAD; e += NALL) {
    int b = e / XPAD, k = e - b * XPAD;
    int idx = x[b * SEQ + t];
    float v = (k < EMBD) ? emb[(size_t)idx * EMBD + k] : 0.f;
    hx0[b * K0LEN + k] = f2bf_bits(v);
  }
}

// Store register-carried h tile (WAR-safe deferred store).
__device__ __forceinline__ void store_hregs(unsigned short* __restrict__ dst,
                                            int stride, int off,
                                            const unsigned short* hreg) {
  const int wave = (blockIdx.x << 3) | (threadIdx.x >> 5);
  const int l = threadIdx.x & 31;
  const int n = wave & 31, m = wave >> 5;
  const int j = (n << 4) + (l & 15);
#pragma unroll
  for (int r = 0; r < 8; ++r) {
    const int b = (m << 4) + r + ((l >> 4) << 3);
    dst[b * stride + off + j] = hreg[r];
  }
}

// One LSTM layer step: gates = act[64xK] * Wg[2048xK]^T ; fused state update.
// Wave (m,n) owns batch rows 16m..16m+15 and hidden cols 16n..16n+15 of all
// four gates -> 4 WMMA accumulators, elementwise done by the producing wave.
__device__ __forceinline__ void lstm_substep(
    const unsigned short* __restrict__ act, int rowlen, int nkt,
    const unsigned short* __restrict__ wg,
    const float* __restrict__ bias,
    float* __restrict__ cst,
    unsigned short* __restrict__ hdst, int hstride, int hoff,  // immediate dest (or null)
    unsigned short* hreg)                                      // register-carried h bits
{
  const int wave = (blockIdx.x << 3) | (threadIdx.x >> 5);
  const int l = threadIdx.x & 31;
  const int n = wave & 31, m = wave >> 5;
  v8f acc0 = {}, acc1 = {}, acc2 = {}, acc3 = {};
  for (int kt = 0; kt < nkt; ++kt) {
    const int k0 = kt << 5;
    v16bf a  = load_A(act, rowlen, m << 4, k0);
    v16bf bI = load_B(wg, rowlen, 0 * HSZ + (n << 4), k0);
    v16bf bF = load_B(wg, rowlen, 1 * HSZ + (n << 4), k0);
    v16bf bG = load_B(wg, rowlen, 2 * HSZ + (n << 4), k0);
    v16bf bO = load_B(wg, rowlen, 3 * HSZ + (n << 4), k0);
    acc0 = __builtin_amdgcn_wmma_f32_16x16x32_bf16(false, a, false, bI, (short)0, acc0, false, false);
    acc1 = __builtin_amdgcn_wmma_f32_16x16x32_bf16(false, a, false, bF, (short)0, acc1, false, false);
    acc2 = __builtin_amdgcn_wmma_f32_16x16x32_bf16(false, a, false, bG, (short)0, acc2, false, false);
    acc3 = __builtin_amdgcn_wmma_f32_16x16x32_bf16(false, a, false, bO, (short)0, acc3, false, false);
  }
  const int j = (n << 4) + (l & 15);
  const float bi = bias[j];
  const float bf = bias[HSZ + j];
  const float bg = bias[2 * HSZ + j];
  const float bo = bias[3 * HSZ + j];
#pragma unroll
  for (int r = 0; r < 8; ++r) {            // C/D layout: lane&15 = N, M = r + 8*(l>>4)
    const int b = (m << 4) + r + ((l >> 4) << 3);
    float iv = sigm_(acc0[r] + bi);
    float fv = sigm_(acc1[r] + bf);
    float gv = tanh_(acc2[r] + bg);
    float ov = sigm_(acc3[r] + bo);
    float c  = fv * cst[b * HSZ + j] + iv * gv;
    cst[b * HSZ + j] = c;
    unsigned short hb = f2bf_bits(ov * tanh_(c));
    hreg[r] = hb;
    if (hdst) hdst[b * hstride + hoff + j] = hb;
  }
}

// Pre-pack: Wg0[2048][832] = [W_ih0 | pad | W_hh0] bf16, Wg1[2048][1024],
// bias = b_ih + b_hh, zero the barrier.
__global__ __launch_bounds__(256) void pack_params(
    const float* __restrict__ Wih0, const float* __restrict__ Whh0,
    const float* __restrict__ bih0, const float* __restrict__ bhh0,
    const float* __restrict__ Wih1, const float* __restrict__ Whh1,
    const float* __restrict__ bih1, const float* __restrict__ bhh1,
    unsigned short* __restrict__ wg0, unsigned short* __restrict__ wg1,
    float* __restrict__ bias0, float* __restrict__ bias1, unsigned* bar)
{
  const int tid = blockIdx.x * blockDim.x + threadIdx.x;
  const int nth = gridDim.x * blockDim.x;
  for (int e = tid; e < G4H * K0LEN; e += nth) {
    int r = e / K0LEN, k = e - r * K0LEN;
    float v = (k < EMBD) ? Wih0[r * EMBD + k]
            : (k < XPAD) ? 0.f
                         : Whh0[r * HSZ + (k - XPAD)];
    wg0[e] = f2bf_bits(v);
  }
  for (int e = tid; e < G4H * K1LEN; e += nth) {
    int r = e >> 10, k = e & 1023;
    float v = (k < HSZ) ? Wih1[r * HSZ + k] : Whh1[r * HSZ + (k - HSZ)];
    wg1[e] = f2bf_bits(v);
  }
  for (int e = tid; e < G4H; e += nth) {
    bias0[e] = bih0[e] + bhh0[e];
    bias1[e] = bih1[e] + bhh1[e];
  }
  if (tid < 2) bar[tid] = 0u;
}

__global__ __launch_bounds__(NTHR) void lstm_persistent(
    const int* __restrict__ x, const float* __restrict__ emb,
    const unsigned short* __restrict__ wg0, const unsigned short* __restrict__ wg1,
    const float* __restrict__ bias0, const float* __restrict__ bias1,
    const float* __restrict__ Wout, const float* __restrict__ bout,
    unsigned short* __restrict__ hx0,   // [64][832]  = [x_t pad | h0]
    unsigned short* __restrict__ hx1,   // [64][1024] = [h0 (L1 input) | h1]
    float* __restrict__ c0, float* __restrict__ c1,
    unsigned* bar, float* __restrict__ out)
{
  const int tid = blockIdx.x * NTHR + threadIdx.x;

  // ---- init: h0=h1=c0=c1=0, gather x_0 ----
  for (int e = tid; e < BATCH * HSZ; e += NALL) {
    int b = e >> 9, k = e & (HSZ - 1);
    hx0[b * K0LEN + XPAD + k] = 0;
    c0[e] = 0.f;
    c1[e] = 0.f;
  }
  for (int e = tid; e < BATCH * K1LEN; e += NALL) hx1[e] = 0;
  gather_x(tid, 0, x, emb, hx0);
  gbar(bar);

  unsigned short h0reg[8], h1reg[8];
#pragma unroll 1
  for (int t = 0; t < SEQ; ++t) {
    // phase 0: readers touch hx0 only -> safe to flush last step's h1 into hx1.
    if (t > 0) store_hregs(hx1, K1LEN, HSZ, h1reg);
    lstm_substep(hx0, K0LEN, K0LEN / 32, wg0, bias0, c0,
                 hx1, K1LEN, 0, h0reg);          // h0 -> L1 input now, keep in regs
    gbar(bar);
    // phase 1: readers touch hx1 only -> safe to flush h0 into hx0 h-slot.
    store_hregs(hx0, K0LEN, XPAD, h0reg);
    lstm_substep(hx1, K1LEN, K1LEN / 32, wg1, bias1, c1,
                 nullptr, 0, 0, h1reg);          // h1 carried in regs
    if (t + 1 < SEQ) gather_x(tid, t + 1, x, emb, hx0);
    gbar(bar);
  }

  // ---- epilogue: out = c1 @ W_out^T + b_out  (64x10, trivially small) ----
  if (blockIdx.x == 0) {
    for (int e = threadIdx.x; e < BATCH * NLBL; e += NTHR) {
      int b = e / NLBL, o = e - b * NLBL;
      float s = bout[o];
      for (int k = 0; k < HSZ; ++k) s += c1[b * HSZ + k] * Wout[o * HSZ + k];
      out[e] = s;
    }
  }
}

// ---------------- workspace layout (all 16B aligned) ----------------
static constexpr size_t OFF_WG0 = 0;
static constexpr size_t SZ_WG0  = (size_t)G4H * K0LEN * 2;
static constexpr size_t OFF_WG1 = OFF_WG0 + SZ_WG0;
static constexpr size_t SZ_WG1  = (size_t)G4H * K1LEN * 2;
static constexpr size_t OFF_B0  = OFF_WG1 + SZ_WG1;
static constexpr size_t OFF_B1  = OFF_B0 + (size_t)G4H * 4;
static constexpr size_t OFF_HX0 = OFF_B1 + (size_t)G4H * 4;
static constexpr size_t OFF_HX1 = OFF_HX0 + (size_t)BATCH * K0LEN * 2;
static constexpr size_t OFF_C0  = OFF_HX1 + (size_t)BATCH * K1LEN * 2;
static constexpr size_t OFF_C1  = OFF_C0 + (size_t)BATCH * HSZ * 4;
static constexpr size_t OFF_BAR = OFF_C1 + (size_t)BATCH * HSZ * 4;

extern "C" void kernel_launch(void* const* d_in, const int* in_sizes, int n_in,
                              void* d_out, int out_size, void* d_ws, size_t ws_size,
                              hipStream_t stream) {
  (void)in_sizes; (void)n_in; (void)out_size; (void)ws_size;
  const int*   x    = (const int*)d_in[0];
  const float* emb  = (const float*)d_in[1];
  const float* Wih0 = (const float*)d_in[2];
  const float* Whh0 = (const float*)d_in[3];
  const float* bih0 = (const float*)d_in[4];
  const float* bhh0 = (const float*)d_in[5];
  const float* Wih1 = (const float*)d_in[6];
  const float* Whh1 = (const float*)d_in[7];
  const float* bih1 = (const float*)d_in[8];
  const float* bhh1 = (const float*)d_in[9];
  const float* Wout = (const float*)d_in[10];
  const float* bout = (const float*)d_in[11];
  float* out = (float*)d_out;

  char* ws = (char*)d_ws;
  unsigned short* wg0   = (unsigned short*)(ws + OFF_WG0);
  unsigned short* wg1   = (unsigned short*)(ws + OFF_WG1);
  float*          bias0 = (float*)(ws + OFF_B0);
  float*          bias1 = (float*)(ws + OFF_B1);
  unsigned short* hx0   = (unsigned short*)(ws + OFF_HX0);
  unsigned short* hx1   = (unsigned short*)(ws + OFF_HX1);
  float*          c0    = (float*)(ws + OFF_C0);
  float*          c1    = (float*)(ws + OFF_C1);
  unsigned*       bar   = (unsigned*)(ws + OFF_BAR);

  pack_params<<<dim3(256), dim3(256), 0, stream>>>(
      Wih0, Whh0, bih0, bhh0, Wih1, Whh1, bih1, bhh1,
      wg0, wg1, bias0, bias1, bar);

  lstm_persistent<<<dim3(NBLK), dim3(NTHR), 0, stream>>>(
      x, emb, wg0, wg1, bias0, bias1, Wout, bout,
      hx0, hx1, c0, c1, bar, out);
}
